// Decoder_74191265071847
// MI455X (gfx1250) — matrix operand order repaired and checked by reading, not compiled, not run on hardware
//
#include <hip/hip_runtime.h>
#include <hip/hip_bf16.h>
#include <stdint.h>

// ---------------------------------------------------------------------------
// Problem constants (from reference): V=32000, WV=300, HID=1024, N=16, T=256
// ---------------------------------------------------------------------------
#define VSZ   32000
#define WV    300
#define HID   1024
#define NB    16
#define TT    256
#define KIN   (HID + WV)        // 1324
#define KINP  1344              // padded to multiple of 32
#define G4H   (4 * HID)         // 4096
#define ROWS  (TT * NB)         // 4096  (row r = t*16 + n)

typedef __attribute__((ext_vector_type(16))) __bf16 v16bf;
typedef __attribute__((ext_vector_type(8)))  float  v8f;

union Frag {
    v16bf v;
    uint4 u[2];
};

static __device__ __forceinline__ uint16_t f2bf(float f) {
    union { float f; uint32_t u; } x;
    x.f = f;
    uint32_t r = x.u + 0x7FFFu + ((x.u >> 16) & 1u);   // round-to-nearest-even
    return (uint16_t)(r >> 16);
}

// ---------------------------------------------------------------------------
// Prep: X[r, c] = [ encoded[n, 0..1023] | emb[y[n,t], 0..299] | zero-pad ]
// stored bf16 row-major with lda = KINP.  r = t*16 + n.
// ---------------------------------------------------------------------------
__global__ __launch_bounds__(256) void build_x_kernel(
    const int* __restrict__ y, const float* __restrict__ encoded,
    const float* __restrict__ emb, uint16_t* __restrict__ Xbf)
{
    int idx = blockIdx.x * 256 + threadIdx.x;
    if (idx >= ROWS * KINP) return;
    int r = idx / KINP, c = idx % KINP;
    int t = r >> 4, n = r & 15;
    float v;
    if (c < HID) {
        v = encoded[n * HID + c];
    } else if (c < KIN) {
        int tok = y[n * TT + t];
        v = emb[(size_t)tok * WV + (c - HID)];
    } else {
        v = 0.0f;
    }
    Xbf[idx] = f2bf(v);
}

// UT[j, k] = bf16(U[k, j]);  U is [1024, 4096], UT is [4096, 1024]
__global__ __launch_bounds__(256) void transpose_u_kernel(
    const float* __restrict__ U, uint16_t* __restrict__ UT)
{
    int idx = blockIdx.x * 256 + threadIdx.x;
    if (idx >= G4H * HID) return;
    int j = idx / HID, k = idx % HID;
    UT[idx] = f2bf(U[(size_t)k * G4H + j]);
}

// ---------------------------------------------------------------------------
// Generic bf16 WMMA GEMM:  C[M=4096, Ntot] = A_bf16[M, Kpad] * B_f32[Kreal, Ntot] + bias
// Block: 256 threads (8 waves), tile 128(M) x 64(N).  Wave w does rows w*16..+15.
// A tile is copied global->LDS with gfx1250 async-LDS DMA ops (ASYNCcnt),
// overlapping the VALU f32->bf16 conversion+transpose of the B tile.
// Fragments are built with two 128-bit ds loads in a contiguous-k (k = 16*half + s, same for A and B) lane
// layout.  remap=1: output row (t*16+n) stored at row (n*256+t) ([N,T,V]).
// ---------------------------------------------------------------------------
__global__ __launch_bounds__(256) void gemm_bf16_kernel(
    const uint16_t* __restrict__ A, int lda,
    const float* __restrict__ B, int ldb,
    const float* __restrict__ bias,
    float* __restrict__ C, int ldc,
    int Kreal, int Kpad, int remap)
{
    __shared__ uint16_t sA[128 * 40];   // row stride 40 el (80 B, 16B-aligned, bank-skewed)
    __shared__ uint16_t sB[64 * 40];    // transposed: sB[col][k], stride 40 el

    const int tid  = threadIdx.x;
    const int wave = tid >> 5;
    const int lane = tid & 31;
    const int ln   = lane & 15;   // col / row-within-tile selector
    const int ks   = lane >> 4;   // K-half selector per ISA lane layout

    const int row0 = blockIdx.y * 128;
    const int col0 = blockIdx.x * 64;

    v8f acc[4] = {};

    for (int k0 = 0; k0 < Kpad; k0 += 32) {
        // ---- stage A chunk [128 x 32] bf16: async global->LDS DMA ----
        // Each thread copies 32 contiguous bytes.  INST_OFFSET is added to both
        // the LDS and memory addresses, so one (lds, vaddr) pair serves both
        // 16-byte halves.  Tracked by ASYNCcnt.
        {
            int ar = tid >> 1, half = tid & 1;
            const uint16_t* src = A + (size_t)(row0 + ar) * lda + k0 + half * 16;
            uint32_t lds0 = (uint32_t)(uintptr_t)(&sA[ar * 40 + half * 16]);
            asm volatile(
                "global_load_async_to_lds_b128 %0, %1, off\n\t"
                "global_load_async_to_lds_b128 %0, %1, off offset:16"
                :: "v"(lds0), "v"(src) : "memory");
        }
        // ---- stage B chunk [32 x 64] f32 -> bf16, transposed into sB[col][k] ----
        {
            int kk = tid >> 3, cg = (tid & 7) << 3;
            int krow = k0 + kk;
            float tmp[8];
            if (krow < Kreal) {
                const float* src = B + (size_t)krow * ldb + col0 + cg;
                float4 f0 = ((const float4*)src)[0];
                float4 f1 = ((const float4*)src)[1];
                tmp[0] = f0.x; tmp[1] = f0.y; tmp[2] = f0.z; tmp[3] = f0.w;
                tmp[4] = f1.x; tmp[5] = f1.y; tmp[6] = f1.z; tmp[7] = f1.w;
                if (krow + 32 < Kreal)
                    __builtin_prefetch(B + (size_t)(krow + 32) * ldb + col0 + cg, 0, 3);
            } else {
#pragma unroll
                for (int i = 0; i < 8; ++i) tmp[i] = 0.0f;
            }
#pragma unroll
            for (int i = 0; i < 8; ++i) sB[(cg + i) * 40 + kk] = f2bf(tmp[i]);
        }
        // async A copy must land in LDS before the tile is published
        asm volatile("s_wait_asynccnt 0" ::: "memory");
        __syncthreads();

        // ---- fragments + WMMA ----
        Frag a;
        {
            const uint16_t* ap = &sA[(wave * 16 + ln) * 40 + ks * 16];
            a.u[0] = ((const uint4*)ap)[0];          // K 0..7   (ks=0) / 16..23 (ks=1)
            a.u[1] = *(const uint4*)(ap + 8);        // K 8..15  (ks=0) / 24..31 (ks=1)
        }
#pragma unroll
        for (int nt = 0; nt < 4; ++nt) {
            Frag b;
            const uint16_t* bp = &sB[(nt * 16 + ln) * 40 + ks * 16];
            b.u[0] = ((const uint4*)bp)[0];          // K 0..15  (ks=0) / 16..31 (ks=1)
            b.u[1] = ((const uint4*)bp)[1];
            acc[nt] = __builtin_amdgcn_wmma_f32_16x16x32_bf16(
                false, a.v, false, b.v, (short)0, acc[nt], false, false);
        }
        __syncthreads();
    }

    // ---- writeback: acc VGPR r, lane -> (M = r + ks*8, Ncol = ln) ----
#pragma unroll
    for (int nt = 0; nt < 4; ++nt) {
        int gc = col0 + nt * 16 + ln;
        float bv = bias ? bias[gc] : 0.0f;
#pragma unroll
        for (int r = 0; r < 8; ++r) {
            int gr = row0 + wave * 16 + r + ks * 8;
            int orow = remap ? ((gr & 15) * TT + (gr >> 4)) : gr;
            C[(size_t)orow * ldc + gc] = acc[nt][r] + bv;
        }
    }
}

// ---------------------------------------------------------------------------
// Persistent single-workgroup LSTM recurrence (one layer, 256 steps).
// 1024 threads = 32 wave32 on one WGP.  h kept in LDS as bf16 (row stride
// 1032 el = 2064 B to break bank conflicts), c kept in registers.
// Each wave owns 32 h-columns: per step it computes the 4 gate tiles for its
// 2 column-tiles (two passes of 4 v8f accumulators) with WMMA against U^T.
//   z[16 x 4096] = Zpre[t] + h_{t-1} @ U         (Zpre holds x@W + b)
// ---------------------------------------------------------------------------
__global__ __launch_bounds__(1024) void lstm_layer_kernel(
    const float* __restrict__ Zpre,        // [ROWS, 4096], row r = t*16+n
    const uint16_t* __restrict__ UT,       // [4096, 1024] bf16, UT[j][k] = U[k][j]
    uint16_t* __restrict__ hs_out,         // [ROWS, 1024] bf16
    const float* __restrict__ h_init,      // [16, 1024] or nullptr (zeros)
    const float* __restrict__ c_init,      // [16, 1024] or nullptr (zeros)
    float* __restrict__ h_fin,             // [16, 1024] f32
    float* __restrict__ c_fin)             // [16, 1024] f32
{
    __shared__ uint16_t hbuf[16 * 1032];

    const int tid  = threadIdx.x;
    const int wave = tid >> 5;
    const int lane = tid & 31;
    const int ln   = lane & 15;
    const int ks   = lane >> 4;
    const int hc0  = wave * 32;            // this wave's h-column base

    // init h in LDS
    for (int i = tid; i < 16 * HID; i += 1024) {
        int n = i >> 10, j = i & 1023;
        float v = h_init ? h_init[i] : 0.0f;
        hbuf[n * 1032 + j] = f2bf(v);
    }
    // init c in registers: element (r, lane) -> n = r + ks*8, col = hc0 + s*16 + ln
    float cacc[2][8];
#pragma unroll
    for (int s = 0; s < 2; ++s)
#pragma unroll
        for (int r = 0; r < 8; ++r) {
            int n = r + ks * 8, col = hc0 + s * 16 + ln;
            cacc[s][r] = c_init ? c_init[n * HID + col] : 0.0f;
        }
    __syncthreads();

    for (int t = 0; t < TT; ++t) {
        float hnew[2][8];

        for (int s = 0; s < 2; ++s) {      // two passes keep live v8f accums at 4
            v8f z[4] = {};
            for (int k = 0; k < HID; k += 32) {
                Frag a;
                const uint16_t* ap = &hbuf[ln * 1032 + k + ks * 16];
                a.u[0] = ((const uint4*)ap)[0];
                a.u[1] = *(const uint4*)(ap + 8);
#pragma unroll
                for (int g = 0; g < 4; ++g) {
                    int j = g * HID + hc0 + s * 16 + ln;    // U^T row = gate column
                    const uint16_t* bp = UT + (size_t)j * HID + k + ks * 16;
                    Frag b;
                    b.u[0] = ((const uint4*)bp)[0];
                    b.u[1] = ((const uint4*)bp)[1];
                    z[g] = __builtin_amdgcn_wmma_f32_16x16x32_bf16(
                        false, a.v, false, b.v, (short)0, z[g], false, false);
                }
            }
            // gates (Keras order i, f, g, o), fp32 VALU
#pragma unroll
            for (int r = 0; r < 8; ++r) {
                int n = r + ks * 8;
                int col = hc0 + s * 16 + ln;
                const float* zp = Zpre + (size_t)(t * 16 + n) * G4H;
                float zi = z[0][r] + zp[col];
                float zf = z[1][r] + zp[HID + col];
                float zg = z[2][r] + zp[2 * HID + col];
                float zo = z[3][r] + zp[3 * HID + col];
                float ig = 1.0f / (1.0f + __expf(-zi));
                float fg = 1.0f / (1.0f + __expf(-zf));
                float gg = tanhf(zg);
                float og = 1.0f / (1.0f + __expf(-zo));
                float c  = fg * cacc[s][r] + ig * gg;
                cacc[s][r] = c;
                hnew[s][r] = og * tanhf(c);
            }
        }

        __syncthreads();                   // all waves finished reading h_{t-1}
#pragma unroll
        for (int s = 0; s < 2; ++s)
#pragma unroll
            for (int r = 0; r < 8; ++r) {
                int n = r + ks * 8;
                int col = hc0 + s * 16 + ln;
                uint16_t hb = f2bf(hnew[s][r]);
                hbuf[n * 1032 + col] = hb;
                hs_out[(size_t)(t * 16 + n) * HID + col] = hb;
            }
        if (t == TT - 1) {
#pragma unroll
            for (int s = 0; s < 2; ++s)
#pragma unroll
                for (int r = 0; r < 8; ++r) {
                    int n = r + ks * 8;
                    int col = hc0 + s * 16 + ln;
                    h_fin[n * HID + col] = hnew[s][r];
                    c_fin[n * HID + col] = cacc[s][r];
                }
        }
        __syncthreads();                   // h_t visible before next step reads
    }
}

// ---------------------------------------------------------------------------
// Launch
// ---------------------------------------------------------------------------
extern "C" void kernel_launch(void* const* d_in, const int* in_sizes, int n_in,
                              void* d_out, int out_size, void* d_ws, size_t ws_size,
                              hipStream_t stream) {
    const int*   y       = (const int*)  d_in[0];
    const float* encoded = (const float*)d_in[1];
    const float* emb     = (const float*)d_in[2];
    const float* W1      = (const float*)d_in[3];
    const float* U1      = (const float*)d_in[4];
    const float* b1      = (const float*)d_in[5];
    const float* W2      = (const float*)d_in[6];
    const float* U2      = (const float*)d_in[7];
    const float* b2      = (const float*)d_in[8];
    const float* Wd      = (const float*)d_in[9];
    const float* bd      = (const float*)d_in[10];
    (void)in_sizes; (void)n_in; (void)out_size; (void)ws_size;

    // workspace layout (bytes, 256-aligned)
    char* ws = (char*)d_ws;
    uint16_t* Xbf   = (uint16_t*)(ws + 0);                       // 4096*1344*2 = 11,010,048
    float*    Zpre  = (float*)   (ws + 11010048);                // 4096*4096*4 = 67,108,864 (reused L1 then L2)
    uint16_t* hs1bf = (uint16_t*)(ws + 78118912);                // 4096*1024*2 =  8,388,608
    uint16_t* hs2bf = (uint16_t*)(ws + 86507520);                //               8,388,608
    uint16_t* U1T   = (uint16_t*)(ws + 94896128);                //               8,388,608
    uint16_t* U2T   = (uint16_t*)(ws + 103284736);               //               8,388,608
    float*    h1    = (float*)   (ws + 111673344);               // 16*1024*4 = 65,536
    float*    c1    = (float*)   (ws + 111738880);               // 65,536   -> total ~106.6 MB

    float* logits = (float*)d_out;                               // [N, T, V]
    float* h2     = logits + (size_t)NB * TT * VSZ;              // [16, 1024]
    float* c2     = h2 + NB * HID;                               // [16, 1024]

    // 1) build bf16 inputs
    build_x_kernel<<<(ROWS * KINP + 255) / 256, 256, 0, stream>>>(y, encoded, emb, Xbf);
    transpose_u_kernel<<<(G4H * HID) / 256, 256, 0, stream>>>(U1, U1T);
    transpose_u_kernel<<<(G4H * HID) / 256, 256, 0, stream>>>(U2, U2T);

    // 2) Z1pre = X @ W1 + b1            (K 1324 padded to 1344)
    gemm_bf16_kernel<<<dim3(G4H / 64, ROWS / 128), 256, 0, stream>>>(
        Xbf, KINP, W1, G4H, b1, Zpre, G4H, KIN, KINP, 0);

    // 3) layer-1 recurrence (h0 = c0 = 0)
    lstm_layer_kernel<<<1, 1024, 0, stream>>>(Zpre, U1T, hs1bf,
                                              nullptr, nullptr, h1, c1);

    // 4) Z2pre = hs1 @ W2 + b2          (reuses Zpre buffer)
    gemm_bf16_kernel<<<dim3(G4H / 64, ROWS / 128), 256, 0, stream>>>(
        hs1bf, HID, W2, G4H, b2, Zpre, G4H, HID, HID, 0);

    // 5) layer-2 recurrence (init = layer-1 final state); finals -> d_out tail
    lstm_layer_kernel<<<1, 1024, 0, stream>>>(Zpre, U2T, hs2bf, h1, c1, h2, c2);

    // 6) logits = hs2 @ Wd + bd, rows remapped (t*16+n) -> (n*256+t)
    gemm_bf16_kernel<<<dim3(VSZ / 64, ROWS / 128), 256, 0, stream>>>(
        hs2bf, HID, Wd, VSZ, bd, logits, VSZ, HID, HID, 1);
}
